// multimodes_actor_70420283785766
// MI455X (gfx1250) — matrix operand order repaired and checked by reading, not compiled, not run on hardware
//
#include <hip/hip_runtime.h>

// ---------------------------------------------------------------------------
// MI455X (gfx1250) fused GCN stack.
// Strategy: convert adjacency fp32->bf16 once (L2-resident afterwards),
// store X@W transposed [F][K] bf16, aggregate with v_wmma_f32_16x16x32_bf16.
// ---------------------------------------------------------------------------

typedef __bf16 bf16;
typedef __attribute__((ext_vector_type(8)))  bf16  v8bf;
typedef __attribute__((ext_vector_type(16))) bf16  v16bf;
typedef __attribute__((ext_vector_type(8)))  float v8f;

__device__ __forceinline__ unsigned short f2bf(float f) {
    union { float f; unsigned int u; } c; c.f = f;
    unsigned int u = c.u;
    u += 0x7fffu + ((u >> 16) & 1u);          // round-to-nearest-even
    return (unsigned short)(u >> 16);
}

// ---------------- fp32 -> bf16 bulk convert (vectorized) -------------------
__global__ void k_cvt_bf16(const float* __restrict__ s, unsigned short* __restrict__ d, long n4) {
    long i = (long)blockIdx.x * blockDim.x + threadIdx.x;
    long stride = (long)gridDim.x * blockDim.x;
    for (; i < n4; i += stride) {
        float4 v = ((const float4*)s)[i];
        ushort4 o;
        o.x = f2bf(v.x); o.y = f2bf(v.y); o.z = f2bf(v.z); o.w = f2bf(v.w);
        ((ushort4*)d)[i] = o;
    }
}

__global__ void k_fill_u16(unsigned short* __restrict__ d, long n, unsigned short v) {
    long i = (long)blockIdx.x * blockDim.x + threadIdx.x;
    long stride = (long)gridDim.x * blockDim.x;
    for (; i < n; i += stride) d[i] = v;
}

__global__ void k_fill_f32(float* __restrict__ d, int n, float v) {
    int i = blockIdx.x * blockDim.x + threadIdx.x;
    if (i < n) d[i] = v;
}

__global__ void k_copy_f32(float* __restrict__ d, const float* __restrict__ s, int n) {
    int i = blockIdx.x * blockDim.x + threadIdx.x;
    if (i < n) d[i] = s[i];
}

// ------------- X@W with transposed bf16 output T[b][f][k] ------------------
// X: fp32 [B][Nr][ldx], uses columns [xoff, xoff+Cin). Output T column-major
// over features so the WMMA B-fragment is a contiguous K-run per lane.
__global__ void k_xform_T(const float* __restrict__ X, int ldx, int xoff,
                          const float* __restrict__ W, int Cin, int Cout,
                          unsigned short* __restrict__ T, int foff, int Ftot,
                          int Nr, int total) {
    int idx = blockIdx.x * blockDim.x + threadIdx.x;
    if (idx >= total) return;
    int n = idx % Nr;
    int f = (idx / Nr) % Cout;
    int b = idx / (Nr * Cout);
    const float* x = X + ((size_t)b * Nr + n) * ldx + xoff;
    float acc = 0.f;
    for (int c = 0; c < Cin; ++c) acc += x[c] * W[c * Cout + f];
    T[((size_t)b * Ftot + foff + f) * Nr + n] = f2bf(acc);
}

// ---------------- WMMA aggregation: C = act(A @ T^T + bias) ----------------
__device__ __forceinline__ v16bf cat16(v8bf lo, v8bf hi) {
    return __builtin_shufflevector(lo, hi, 0,1,2,3,4,5,6,7,8,9,10,11,12,13,14,15);
}

__device__ __forceinline__ void store_tile(float* __restrict__ C, const float* __restrict__ bias,
                                           int F, int row_base, int col_base,
                                           const v8f& acc, int act, int lane) {
    int n  = col_base + (lane & 15);
    int m0 = row_base + ((lane >> 4) << 3);
    float bv = bias[n];
#pragma unroll
    for (int v = 0; v < 8; ++v) {
        float x = acc[v] + bv;
        x = act ? (1.f / (1.f + __expf(-x))) : fmaxf(x, 0.f);
        C[(size_t)(m0 + v) * F + n] = x;
    }
}

// A: bf16 [b][M][M] row-major. T: bf16 [b][F][M]. C: fp32 [b][M][F].
// block = 256 threads = 8 waves in a 4(row) x 2(col) arrangement;
// block tile = 128 x 64, wave tile = 32 x 32 (4 accumulators).
__global__ __launch_bounds__(256) void k_gcn_wmma(
        const unsigned short* __restrict__ Abf, const unsigned short* __restrict__ Tbf,
        const float* __restrict__ bias, float* __restrict__ C,
        int M, int F, int act)
{
    const int lane = threadIdx.x & 31;
    const int w    = threadIdx.x >> 5;
    const int wr   = w & 3;
    const int wc   = w >> 2;
    const int b    = blockIdx.z;
    const int row0 = blockIdx.x * 128 + wr * 32;
    const int col0 = blockIdx.y * 64  + wc * 32;

    const bf16* A = (const bf16*)(Abf + (size_t)b * M * M);
    const bf16* T = (const bf16*)(Tbf + (size_t)b * F * M);
    float*      Cb = C + (size_t)b * M * F;

    const bool ok0 = (col0 + 16) <= F;              // guard for padded F (F=16 layer)
    const bool ok1 = (col0 + 32) <= F;
    const int  c0  = ok0 ? col0        : 0;
    const int  c1  = ok1 ? (col0 + 16) : 0;

    const int lr    = lane & 15;
    const int klo_a = (lane >> 4) << 3;             // A frag: chunks at +0/+16 (or +8/+24)
    const int klo_b = (lane >> 4) << 3;             // B frag: chunks at +0/+16 (or +8/+24)

    const bf16* pa0 = A + (size_t)(row0      + lr) * M + klo_a;
    const bf16* pa1 = A + (size_t)(row0 + 16 + lr) * M + klo_a;
    const bf16* pb0 = T + (size_t)(c0 + lr) * M + klo_b;
    const bf16* pb1 = T + (size_t)(c1 + lr) * M + klo_b;

    v8f acc00 = {}, acc01 = {}, acc10 = {}, acc11 = {};

    for (int k = 0; k < M; k += 32) {
        if (k + 32 < M) {                            // global_prefetch_b8 on next K-slab
            __builtin_prefetch(pa0 + k + 32, 0, 0);
            __builtin_prefetch(pa1 + k + 32, 0, 0);
        }
        v16bf a0 = cat16(*(const v8bf*)(pa0 + k), *(const v8bf*)(pa0 + k + 16));
        v16bf a1 = cat16(*(const v8bf*)(pa1 + k), *(const v8bf*)(pa1 + k + 16));
        v16bf b0 = cat16(*(const v8bf*)(pb0 + k), *(const v8bf*)(pb0 + k + 16));
        v16bf b1 = cat16(*(const v8bf*)(pb1 + k), *(const v8bf*)(pb1 + k + 16));
        acc00 = __builtin_amdgcn_wmma_f32_16x16x32_bf16(false, a0, false, b0, (short)0, acc00, false, false);
        acc01 = __builtin_amdgcn_wmma_f32_16x16x32_bf16(false, a0, false, b1, (short)0, acc01, false, false);
        acc10 = __builtin_amdgcn_wmma_f32_16x16x32_bf16(false, a1, false, b0, (short)0, acc10, false, false);
        acc11 = __builtin_amdgcn_wmma_f32_16x16x32_bf16(false, a1, false, b1, (short)0, acc11, false, false);
    }

    if (ok0) store_tile(Cb, bias, F, row0,      c0, acc00, act, lane);
    if (ok1) store_tile(Cb, bias, F, row0,      c1, acc01, act, lane);
    if (ok0) store_tile(Cb, bias, F, row0 + 16, c0, acc10, act, lane);
    if (ok1) store_tile(Cb, bias, F, row0 + 16, c1, acc11, act, lane);
}

// --------------------------- small glue kernels ----------------------------
__global__ void k_pool(const float* __restrict__ P1, float* __restrict__ pooled, int P, int H, int total) {
    int idx = blockIdx.x * blockDim.x + threadIdx.x;
    if (idx >= total) return;
    int h = idx % H, b = idx / H;
    float s = 0.f;
    for (int n = 0; n < P; ++n) s += P1[((size_t)b * P + n) * H + h];
    pooled[idx] = s;
}

// x_1_4[b,i,j] = pooled[b, (i*H+j) // N]  (broadcast-to-[B,H,N] then reshape)
__global__ void k_x14(const float* __restrict__ pooled, float* __restrict__ X14,
                      int N, int H, int total) {
    int idx = blockIdx.x * blockDim.x + threadIdx.x;
    if (idx >= total) return;
    int j = idx % H;
    int i = (idx / H) % N;
    int b = idx / (H * N);
    X14[idx] = pooled[b * H + (i * H + j) / N];
}

// x_3 = Y2n[:, :, 0:64] + Yts + Ycs + Ys + Y2n[:, :, 64:128]
__global__ void k_add5(const float* __restrict__ Y2n, const float* __restrict__ Yts,
                       const float* __restrict__ Ycs, const float* __restrict__ Ys,
                       float* __restrict__ X3, int N, int H, int total) {
    int idx = blockIdx.x * blockDim.x + threadIdx.x;
    if (idx >= total) return;
    int h  = idx % H;
    int bn = idx / H;
    X3[idx] = Y2n[(size_t)bn * 2 * H + h] + Y2n[(size_t)bn * 2 * H + H + h]
            + Yts[idx] + Ycs[idx] + Ys[idx];
}

// S4[b][n][16] (cols 0-3 -> out_1, 4-11 -> out_2) scattered into flat d_out
__global__ void k_scatter(const float* __restrict__ S4, float* __restrict__ out,
                          int N, int total) {
    int idx = blockIdx.x * blockDim.x + threadIdx.x;
    if (idx >= total) return;
    int j  = idx % 12;
    int bn = idx / 12;
    float v = S4[(size_t)bn * 16 + j];
    if (j < 4) out[(size_t)bn * 4 + j] = v;
    else       out[(size_t)(4 /*B*/ * N) * 4 + (size_t)bn * 8 + (j - 4)] = v;
}

// ---------------------------------------------------------------------------
extern "C" void kernel_launch(void* const* d_in, const int* in_sizes, int n_in,
                              void* d_out, int out_size, void* d_ws, size_t ws_size,
                              hipStream_t stream) {
    constexpr int kB = 4, kN = 2048, kP = 512, kH = 64;
    (void)in_sizes; (void)out_size; (void)ws_size;

    const float* x_n    = (const float*)d_in[0];
    const float* A_n    = (const float*)d_in[1];
    const float* A_s    = (const float*)d_in[2];
    const float* A_n_ts = (const float*)d_in[3];
    const float* A_n_cs = (const float*)d_in[4];
    const float* x_p    = (const float*)d_in[5];
    const float* A_p    = (const float*)d_in[6];

    // params in setup_inputs insertion order: (W,b) per layer
    // l1_1 l1_2 l1_3 l1_4 l2_1 l2_2 l2_3 l2_4 l2_5 l3_1 l3_2 l4_1 l4_2
    static const int fin[13]  = {16,16,16,16,64,64,64,64,64,64,64,64,64};
    static const int fout[13] = {64,64,64,64,64,64,64,64,64,64,64, 4, 8};
    const float* Wp[13]; const float* bp[13];
    if (n_in >= 7 + 26) {
        for (int i = 0; i < 13; ++i) {
            Wp[i] = (const float*)d_in[7 + 2 * i];
            bp[i] = (const float*)d_in[7 + 2 * i + 1];
        }
    } else {
        const float* base = (const float*)d_in[7];
        size_t off = 0;
        for (int i = 0; i < 13; ++i) {
            Wp[i] = base + off; off += (size_t)fin[i] * fout[i];
            bp[i] = base + off; off += fout[i];
        }
    }

    // ---- workspace carve ----
    char* wsb = (char*)d_ws; size_t woff = 0;
    auto carve = [&](size_t bytes) -> void* {
        void* p = wsb + woff; woff += (bytes + 255) & ~(size_t)255; return p;
    };
    unsigned short* Anb  = (unsigned short*)carve((size_t)kB * kN * kN * 2);
    unsigned short* Asb  = (unsigned short*)carve((size_t)kB * kN * kN * 2);
    unsigned short* Atsb = (unsigned short*)carve((size_t)kB * kN * kN * 2);
    unsigned short* Acsb = (unsigned short*)carve((size_t)kB * kN * kN * 2);
    unsigned short* Apb  = (unsigned short*)carve((size_t)kB * kP * kP * 2);
    unsigned short* Tbuf = (unsigned short*)carve((size_t)kB * 192 * kN * 2);
    float* biasb  = (float*)carve(192 * 4);
    float* X1     = (float*)carve((size_t)kB * kN * 192 * 4);
    float* P1     = (float*)carve((size_t)kB * kP * kH * 4);
    float* pooled = (float*)carve((size_t)kB * kH * 4);
    float* X14    = (float*)carve((size_t)kB * kN * kH * 4);
    float* Y2n    = (float*)carve((size_t)kB * kN * 128 * 4);
    float* Yts    = (float*)carve((size_t)kB * kN * kH * 4);
    float* Ycs    = (float*)carve((size_t)kB * kN * kH * 4);
    float* Ys     = (float*)carve((size_t)kB * kN * kH * 4);
    float* X3     = (float*)carve((size_t)kB * kN * kH * 4);
    float* X31    = (float*)carve((size_t)kB * kN * kH * 4);
    float* X32    = (float*)carve((size_t)kB * kN * kH * 4);
    float* S4     = (float*)carve((size_t)kB * kN * 16 * 4);

    auto cvt = [&](const float* s, unsigned short* d, size_t n) {
        long n4 = (long)(n / 4);
        int blk = (int)((n4 + 255) / 256);
        k_cvt_bf16<<<blk, 256, 0, stream>>>(s, d, n4);
    };
    auto xf = [&](const float* X, int ldx, int xoff, const float* W, int Cin, int Cout,
                  unsigned short* T, int foff, int Ftot, int Nr) {
        int total = kB * Nr * Cout;
        k_xform_T<<<(total + 255) / 256, 256, 0, stream>>>(X, ldx, xoff, W, Cin, Cout,
                                                           T, foff, Ftot, Nr, total);
    };
    auto gemm = [&](const unsigned short* A, const unsigned short* T, const float* bias,
                    float* C, int M, int F, int act) {
        dim3 grid(M / 128, (F + 63) / 64, kB);
        k_gcn_wmma<<<grid, 256, 0, stream>>>(A, T, bias, C, M, F, act);
    };
    auto bcopy = [&](float* dst, const float* src, int n) {
        k_copy_f32<<<(n + 63) / 64, 64, 0, stream>>>(dst, src, n);
    };

    // ---- Phase 0: adjacency fp32 -> bf16 (then L2-resident) ----
    cvt(A_n,    Anb,  (size_t)kB * kN * kN);
    cvt(A_s,    Asb,  (size_t)kB * kN * kN);
    cvt(A_n_ts, Atsb, (size_t)kB * kN * kN);
    cvt(A_n_cs, Acsb, (size_t)kB * kN * kN);
    cvt(A_p,    Apb,  (size_t)kB * kP * kP);

    // ---- Layer 1 (fused A_n pass, F=192: l1_1|l1_2|l1_3) ----
    bcopy(biasb +   0, bp[0], 64);
    bcopy(biasb +  64, bp[1], 64);
    bcopy(biasb + 128, bp[2], 64);
    xf(x_n, 16, 0, Wp[0], 16, 64, Tbuf,   0, 192, kN);
    xf(x_n, 16, 0, Wp[1], 16, 64, Tbuf,  64, 192, kN);
    xf(x_n, 16, 0, Wp[2], 16, 64, Tbuf, 128, 192, kN);
    gemm(Anb, Tbuf, biasb, X1, kN, 192, 0);

    // ---- l1_4 on A_p, pool, broadcast-reshape ----
    bcopy(biasb, bp[3], 64);
    xf(x_p, 16, 0, Wp[3], 16, 64, Tbuf, 0, 64, kP);
    gemm(Apb, Tbuf, biasb, P1, kP, 64, 0);
    k_pool<<<1, 256, 0, stream>>>(P1, pooled, kP, kH, kB * kH);
    {
        int total = kB * kN * kH;
        k_x14<<<(total + 255) / 256, 256, 0, stream>>>(pooled, X14, kN, kH, total);
    }

    // ---- Layer 2 group A_n (fused F=128: l2_1 on x_1_1 | l2_5 on x_1_4) ----
    bcopy(biasb +  0, bp[4], 64);
    bcopy(biasb + 64, bp[8], 64);
    xf(X1, 192,   0, Wp[4], 64, 64, Tbuf,  0, 128, kN);
    xf(X14, 64,   0, Wp[8], 64, 64, Tbuf, 64, 128, kN);
    gemm(Anb, Tbuf, biasb, Y2n, kN, 128, 0);

    // ---- Layer 2: ts / cs / s groups (each F=64) ----
    bcopy(biasb, bp[5], 64);
    xf(X1, 192,  64, Wp[5], 64, 64, Tbuf, 0, 64, kN);
    gemm(Atsb, Tbuf, biasb, Yts, kN, 64, 0);

    bcopy(biasb, bp[6], 64);
    xf(X1, 192,  64, Wp[6], 64, 64, Tbuf, 0, 64, kN);
    gemm(Acsb, Tbuf, biasb, Ycs, kN, 64, 0);

    bcopy(biasb, bp[7], 64);
    xf(X1, 192, 128, Wp[7], 64, 64, Tbuf, 0, 64, kN);
    gemm(Asb, Tbuf, biasb, Ys, kN, 64, 0);

    // ---- x_3 = sum of five branches ----
    {
        int total = kB * kN * kH;
        k_add5<<<(total + 255) / 256, 256, 0, stream>>>(Y2n, Yts, Ycs, Ys, X3, kN, kH, total);
    }

    // ---- Layer 3 ----
    bcopy(biasb, bp[9], 64);
    xf(X3, 64, 0, Wp[9], 64, 64, Tbuf, 0, 64, kN);
    gemm(Anb, Tbuf, biasb, X31, kN, 64, 0);

    bcopy(biasb, bp[10], 64);
    xf(X3, 64, 0, Wp[10], 64, 64, Tbuf, 0, 64, kN);
    gemm(Asb, Tbuf, biasb, X32, kN, 64, 0);

    // ---- Layer 4 (fused A_n pass, F padded to 16: l4_1(4)|l4_2(8)|pad(4)) ----
    {
        long tn = (long)kB * 16 * kN;
        k_fill_u16<<<(int)((tn + 255) / 256), 256, 0, stream>>>(Tbuf, tn, (unsigned short)0);
    }
    k_fill_f32<<<1, 32, 0, stream>>>(biasb, 16, 0.f);
    bcopy(biasb + 0, bp[11], 4);
    bcopy(biasb + 4, bp[12], 8);
    xf(X31, 64, 0, Wp[11], 64, 4, Tbuf, 0, 16, kN);
    xf(X32, 64, 0, Wp[12], 64, 8, Tbuf, 4, 16, kN);
    gemm(Anb, Tbuf, biasb, S4, kN, 16, 1 /*sigmoid*/);

    // ---- scatter to d_out: out_1 [B,N,4] then out_2 [B,N,8] ----
    {
        int total = kB * kN * 12;
        k_scatter<<<(total + 255) / 256, 256, 0, stream>>>(S4, (float*)d_out, kN, total);
    }
}
